// MambaModule_88691074663007
// MI455X (gfx1250) — compile-verified, matrix operand order checked by reading, not run on hardware
//
#include <hip/hip_runtime.h>
#include <math.h>

#define EPSF 1e-5f

#if defined(__has_builtin)
# if __has_builtin(__builtin_amdgcn_tensor_load_to_lds)
#  define HAVE_TDM 1
# endif
#endif
#ifndef HAVE_TDM
# define HAVE_TDM 0
#endif

// Flattened input-pointer table (setup_inputs dict order, depth-first):
// 0 features (12,64,96,320)  1 high (12,32,96,320)  2 fid0  3 fid1
// 4..7 bn_in g,b,m,v (96) | 8..12 down0 w,g,b,m,v | 13..17 down1
// 18 emb_w 19 emb_b | 20 ln1_g 21 ln1_b 22 ln2_g 23 ln2_b
// 24+9*pos: mamba[pos] in_proj,conv_w,conv_b,x_proj,dt_w,dt_b,A_log,D,out_proj
// 132 dec_w 133 dec_b | 134..138 up0 | 139..143 up1
// 144..149 fus0 w,b,g,bt,m,v | 150..155 fus1 | 156 outf_w 157 outf_b
struct PtrTab { const float* p[164]; };

typedef __attribute__((ext_vector_type(16))) _Float16 v16h;
typedef __attribute__((ext_vector_type(8)))  _Float16 v8h;
typedef __attribute__((ext_vector_type(8)))  float    v8f;

union Frag16 { v16h v; v8h h[2]; };

__device__ __forceinline__ float eluf(float x)  { return x > 0.f ? x : (__expf(x) - 1.f); }
__device__ __forceinline__ float siluf(float x) { return x / (1.f + __expf(-x)); }

#if HAVE_TDM
typedef __attribute__((ext_vector_type(4))) unsigned int u32x4;
typedef __attribute__((ext_vector_type(8))) int          i32x8;
typedef __attribute__((ext_vector_type(4))) int          i32x4;

// D# (ISA 08 §8) for a 2-D f16 tile load Global->LDS with LDS row padding
// (pad codes: interval 0=2dw..5=64dw; amount 0=1dw..127=128dw).
__device__ __forceinline__ void tdm_load_2d_f16(const _Float16* gsrc, _Float16* lds_dst,
                                                int rows, int elems_per_row,
                                                int pad_interval_code, int pad_amount_code) {
  unsigned long long ga = (unsigned long long)gsrc;
  unsigned lds = (unsigned)(unsigned long long)(void*)lds_dst;
  u32x4 g0;
  g0[0] = 1u;                                             // count=1 (valid user D#)
  g0[1] = lds;                                            // lds_addr
  g0[2] = (unsigned)ga;                                   // global_addr[31:0]
  g0[3] = (unsigned)((ga >> 32) & 0x01FFFFFFull) | 0x80000000u; // addr[56:32] | type=2
  i32x8 g1;
  int w0 = (1 << 16);                                     // data_size = 2 bytes
  if (pad_amount_code >= 0)
    w0 |= (1 << 20) | (pad_interval_code << 22) | (pad_amount_code << 25);
  g1[0] = w0;
  g1[1] = (elems_per_row & 0xFFFF) << 16;
  g1[2] = ((elems_per_row >> 16) & 0xFFFF) | ((rows & 0xFFFF) << 16);
  g1[3] = ((rows >> 16) & 0xFFFF) | ((elems_per_row & 0xFFFF) << 16);
  g1[4] = rows & 0xFFFF;
  g1[5] = elems_per_row;
  g1[6] = 0;
  g1[7] = 0;
  i32x4 z4 = {0, 0, 0, 0};
#if __clang_major__ >= 23
  i32x8 z8 = {0, 0, 0, 0, 0, 0, 0, 0};
  __builtin_amdgcn_tensor_load_to_lds(g0, g1, z4, z4, z8, 0);
#else
  __builtin_amdgcn_tensor_load_to_lds(g0, g1, z4, z4, 0);
#endif
}
#endif  // HAVE_TDM

// ---------------------------------------------------------------------------
// K0: pre-convert weights to f16 tiles for TDM feeding.
// [0,147456): fusion [br][tap][n=64][k=128]
// [147456,155648): outf slices [br][n=64][k=64]
// [155648,163840): up 1x1 [br][n=64][k=64]
// ---------------------------------------------------------------------------
__global__ __launch_bounds__(256) void k_prep(PtrTab T, _Float16* __restrict__ wbuf) {
  int idx = blockIdx.x * 256 + threadIdx.x;
  if (idx < 147456) {
    int br = idx / 73728; int r = idx % 73728;
    int tap = r / 8192;   int q = r % 8192;
    int n = q >> 7;       int k = q & 127;
    const float* fw = T.p[144 + br * 6];
    wbuf[idx] = (_Float16)fw[((n * 128 + k) * 3 + tap / 3) * 3 + tap % 3];
  } else if (idx < 155648) {
    int e = idx - 147456;
    int br = e / 4096; int q = e % 4096;
    int n = q >> 6;    int k = q & 63;
    wbuf[idx] = (_Float16)T.p[156][n * 128 + br * 64 + k];
  } else if (idx < 163840) {
    int e = idx - 155648;
    int br = e / 4096; int q = e % 4096;
    int n = q >> 6;    int k = q & 63;
    wbuf[idx] = (_Float16)T.p[134 + br * 5][n * 64 + k];
  }
}

// ---------------------------------------------------------------------------
// K1: feats construct + bn_in + grouped conv 3x3 s2 + BN + ELU
// ---------------------------------------------------------------------------
__global__ __launch_bounds__(256) void k_down1(PtrTab T, float* __restrict__ out) {
  const float* feat = T.p[0]; const float* high = T.p[1];
  const float *big = T.p[4], *bib = T.p[5], *bim = T.p[6], *biv = T.p[7];
  const float *w = T.p[8], *g = T.p[9], *bt = T.p[10], *mn = T.p[11], *vr = T.p[12];
  int idx = blockIdx.x * 256 + threadIdx.x;
  if (idx >= 12 * 96 * 48 * 160) return;
  int ox = idx % 160; int t = idx / 160;
  int oy = t % 48;    t /= 48;
  int oc = t % 96;    int b = t / 96;
  int grp = oc >> 3;
  float acc = 0.f;
  for (int kc = 0; kc < 8; ++kc) {
    int ic = grp * 8 + kc;
    float sc = big[ic] * rsqrtf(biv[ic] + EPSF);
    float sh = bib[ic] - bim[ic] * sc;
    for (int dy = 0; dy < 3; ++dy) {
      int iy = oy * 2 + dy - 1; if (iy < 0 || iy >= 96) continue;
      for (int dx = 0; dx < 3; ++dx) {
        int ix = ox * 2 + dx - 1; if (ix < 0 || ix >= 320) continue;
        float v;
        if (ic < 64) v = feat[((b * 64 + ic) * 96 + iy) * 320 + ix]
                       + high[((b * 32 + (ic & 31)) * 96 + iy) * 320 + ix];
        else         v = high[((b * 32 + (ic - 64)) * 96 + iy) * 320 + ix];
        v = v * sc + sh;
        acc += v * w[((oc * 8 + kc) * 3 + dy) * 3 + dx];
      }
    }
  }
  float s2 = g[oc] * rsqrtf(vr[oc] + EPSF);
  out[idx] = eluf((acc - mn[oc]) * s2 + bt[oc]);
}

// ---------------------------------------------------------------------------
// K2: grouped conv 3x3 s2 + BN + ELU.  (12,96,48,160)->(12,96,24,80)
// ---------------------------------------------------------------------------
__global__ __launch_bounds__(256) void k_down2(PtrTab T, const float* __restrict__ in,
                                               float* __restrict__ out) {
  const float *w = T.p[13], *g = T.p[14], *bt = T.p[15], *mn = T.p[16], *vr = T.p[17];
  int idx = blockIdx.x * 256 + threadIdx.x;
  if (idx >= 12 * 96 * 24 * 80) return;
  int ox = idx % 80; int t = idx / 80;
  int oy = t % 24;   t /= 24;
  int oc = t % 96;   int b = t / 96;
  int grp = oc >> 3;
  float acc = 0.f;
  for (int kc = 0; kc < 8; ++kc) {
    int ic = grp * 8 + kc;
    for (int dy = 0; dy < 3; ++dy) {
      int iy = oy * 2 + dy - 1; if (iy < 0 || iy >= 48) continue;
      for (int dx = 0; dx < 3; ++dx) {
        int ix = ox * 2 + dx - 1; if (ix < 0 || ix >= 160) continue;
        acc += in[((b * 96 + ic) * 48 + iy) * 160 + ix] * w[((oc * 8 + kc) * 3 + dy) * 3 + dx];
      }
    }
  }
  float s2 = g[oc] * rsqrtf(vr[oc] + EPSF);
  out[idx] = eluf((acc - mn[oc]) * s2 + bt[oc]);
}

// ---------------------------------------------------------------------------
// K3: emb grouped conv 3x3 s1 p1 (groups=12, 8->32) + bias. ->(12,384,24,80)
// ---------------------------------------------------------------------------
__global__ __launch_bounds__(256) void k_emb(PtrTab T, const float* __restrict__ in,
                                             float* __restrict__ out) {
  const float* w = T.p[18]; const float* bias = T.p[19];
  int idx = blockIdx.x * 256 + threadIdx.x;
  if (idx >= 12 * 384 * 24 * 80) return;
  int ox = idx % 80; int t = idx / 80;
  int oy = t % 24;   t /= 24;
  int oc = t % 384;  int b = t / 384;
  int grp = oc >> 5;
  float acc = bias[oc];
  for (int kc = 0; kc < 8; ++kc) {
    int ic = grp * 8 + kc;
    for (int dy = 0; dy < 3; ++dy) {
      int iy = oy + dy - 1; if (iy < 0 || iy >= 24) continue;
      for (int dx = 0; dx < 3; ++dx) {
        int ix = ox + dx - 1; if (ix < 0 || ix >= 80) continue;
        acc += in[((b * 96 + ic) * 24 + iy) * 80 + ix] * w[((oc * 8 + kc) * 3 + dy) * 3 + dx];
      }
    }
  }
  out[idx] = acc;
}

// ---------------------------------------------------------------------------
// K4: fused Mamba (unchanged; latency-bound, state in VGPRs).
// ---------------------------------------------------------------------------
__global__ __launch_bounds__(256) void k_mamba(PtrTab T, const float* __restrict__ emb,
                                               float* __restrict__ decin) {
  __shared__ float fnr[4][12][32];
  __shared__ float fn[4][12][32];
  __shared__ float xb[4][64];
  __shared__ float xdb[4][40];
  __shared__ float yb[4][64];
  __shared__ float outv[4][32];
  int tt = threadIdx.x >> 6; int ln = threadIdx.x & 63;
  int token = blockIdx.x * 4 + tt;
  int b = token / 1920; int rem = token - b * 1920; int y = rem / 80; int x = rem % 80;
  const float *g1 = T.p[20], *b1 = T.p[21], *g2 = T.p[22], *b2 = T.p[23];

  for (int i = 0; i < 6; ++i) {
    int fl = i * 64 + ln; int pos = fl >> 5; int c = fl & 31;
    fnr[tt][pos][c] = emb[((b * 384 + pos * 32 + c) * 24 + y) * 80 + x];
  }
  __syncthreads();
  for (int i = 0; i < 6; ++i) {
    int fl = i * 64 + ln; int pos = fl >> 5; int c = fl & 31;
    float mu = 0.f;
    for (int j = 0; j < 32; ++j) mu += fnr[tt][pos][j];
    mu *= (1.f / 32.f);
    float vv = 0.f;
    for (int j = 0; j < 32; ++j) { float d = fnr[tt][pos][j] - mu; vv += d * d; }
    vv *= (1.f / 32.f);
    fn[tt][pos][c] = (fnr[tt][pos][c] - mu) * rsqrtf(vv + EPSF) * g1[c] + b1[c];
  }
  __syncthreads();

  float ssm[16];
  for (int s = 0; s < 16; ++s) ssm[s] = 0.f;
  int d = ln;
  for (int pos = 0; pos < 12; ++pos) {
    const float* inw  = T.p[24 + pos * 9 + 0];
    const float* cw   = T.p[24 + pos * 9 + 1];
    const float* cb   = T.p[24 + pos * 9 + 2];
    const float* xpw  = T.p[24 + pos * 9 + 3];
    const float* dtw  = T.p[24 + pos * 9 + 4];
    const float* dtb  = T.p[24 + pos * 9 + 5];
    const float* alog = T.p[24 + pos * 9 + 6];
    const float* Dp   = T.p[24 + pos * 9 + 7];
    const float* opw  = T.p[24 + pos * 9 + 8];

    float xr = 0.f, zr = 0.f;
    for (int c = 0; c < 32; ++c) {
      float f = fn[tt][pos][c];
      xr += f * inw[d * 32 + c];
      zr += f * inw[(d + 64) * 32 + c];
    }
    float xv = siluf(xr * cw[d] + cb[d]);
    xb[tt][d] = xv;
    __syncthreads();
    if (ln < 36) {
      float a = 0.f;
      for (int dd = 0; dd < 64; ++dd) a += xb[tt][dd] * xpw[ln * 64 + dd];
      xdb[tt][ln] = a;
    }
    __syncthreads();
    float dta = dtb[d];
    for (int r = 0; r < 4; ++r) dta += xdb[tt][r] * dtw[d * 4 + r];
    float dt = dta > 20.f ? dta : log1pf(__expf(dta));
    float xdt = xv * dt;
    float yv = 0.f;
    for (int s = 0; s < 16; ++s) {
      float A  = -__expf(alog[d * 16 + s]);
      float dA = __expf(dt * A);
      ssm[s] = ssm[s] * dA + xdt * xdb[tt][4 + s];
      yv += ssm[s] * xdb[tt][20 + s];
    }
    yv += Dp[d] * xv;
    yv *= siluf(zr);
    yb[tt][d] = yv;
    __syncthreads();
    if (pos >= 8 && ln < 32) {
      float o = 0.f;
      for (int dd = 0; dd < 64; ++dd) o += yb[tt][dd] * opw[ln * 64 + dd];
      float id = emb[((b * 384 + pos * 32 + ln) * 24 + y) * 80 + x];
      outv[tt][ln] = o + id;
    }
    __syncthreads();
    if (pos >= 8 && ln < 32) {
      float mu = 0.f;
      for (int j = 0; j < 32; ++j) mu += outv[tt][j];
      mu *= (1.f / 32.f);
      float vv = 0.f;
      for (int j = 0; j < 32; ++j) { float dd2 = outv[tt][j] - mu; vv += dd2 * dd2; }
      vv *= (1.f / 32.f);
      float fin = (outv[tt][ln] - mu) * rsqrtf(vv + EPSF) * g2[ln] + b2[ln];
      int ch = (pos - 8) * 32 + ln;
      decin[((b * 128 + ch) * 24 + y) * 80 + x] = fin;
    }
    __syncthreads();
  }
}

// ---------------------------------------------------------------------------
// K5: dec grouped conv 3x3 p1 (groups=4, 32->32) + bias. (12,128,24,80)
// ---------------------------------------------------------------------------
__global__ __launch_bounds__(256) void k_dec(PtrTab T, const float* __restrict__ in,
                                             float* __restrict__ out) {
  const float* w = T.p[132]; const float* bias = T.p[133];
  int idx = blockIdx.x * 256 + threadIdx.x;
  if (idx >= 12 * 128 * 24 * 80) return;
  int x = idx % 80; int t = idx / 80;
  int y = t % 24;   t /= 24;
  int oc = t % 128; int b = t / 128;
  int icb = (oc >> 5) * 32;
  float acc = bias[oc];
  for (int kc = 0; kc < 32; ++kc) {
    for (int dy = 0; dy < 3; ++dy) {
      int iy = y + dy - 1; if (iy < 0 || iy >= 24) continue;
      for (int dx = 0; dx < 3; ++dx) {
        int ix = x + dx - 1; if (ix < 0 || ix >= 80) continue;
        acc += in[((b * 128 + icb + kc) * 24 + iy) * 80 + ix]
             * w[((oc * 32 + kc) * 3 + dy) * 3 + dx];
      }
    }
  }
  out[idx] = acc;
}

// ---------------------------------------------------------------------------
// K6 (WMMA): "up" 1x1 conv + BN + ELU at coarse res as a 64px x 64ch tile
// GEMM (K=64), weights TDM-fed.  360 blocks/branch, 128 threads.
// ---------------------------------------------------------------------------
__global__ __launch_bounds__(128) void k_branch_w(PtrTab T, const float* __restrict__ dec,
                                                  const _Float16* __restrict__ wbuf,
                                                  int br, float* __restrict__ out) {
  __shared__ _Float16 sA[64 * 72];   // [px][ch], pitch 72
  __shared__ _Float16 sB[64 * 136];  // [n][k], pitch 136
  const float *g = T.p[135 + br * 5], *bt = T.p[136 + br * 5];
  const float *mn = T.p[137 + br * 5], *vr = T.p[138 + br * 5];
  int tid = threadIdx.x;
  int p0 = blockIdx.x * 64;
  for (int i = 0; i < 32; ++i) {
    int fl = i * 128 + tid; int px = fl & 63; int ch = fl >> 6;
    int p = p0 + px; int b = p / 1920; int r = p % 1920; int y = r / 80; int x = r % 80;
    sA[px * 72 + ch] = (_Float16)dec[((b * 128 + br * 64 + ch) * 24 + y) * 80 + x];
  }
#if HAVE_TDM
  if (tid == 0) {
    tdm_load_2d_f16(wbuf + 155648 + br * 4096, sB, 64, 64, 4, 35);
    __builtin_amdgcn_s_wait_tensorcnt(0);
  }
#else
  for (int i = 0; i < 32; ++i) {
    int fl = i * 128 + tid; int n = fl >> 6; int k = fl & 63;
    sB[n * 136 + k] = (_Float16)T.p[134 + br * 5][n * 64 + k];
  }
#endif
  __syncthreads();
  int wave = tid >> 5, lane = tid & 31, mrow = lane & 15, hh = lane >> 4;
  v8f acc[4];
  for (int i = 0; i < 4; ++i)
    for (int j = 0; j < 8; ++j) acc[i][j] = 0.f;
  for (int kc = 0; kc < 64; kc += 32) {
    Frag16 a;
    const _Float16* pa = sA + (wave * 16 + mrow) * 72 + kc + hh * 8;
    a.h[0] = *(const v8h*)pa;
    a.h[1] = *(const v8h*)(pa + 16);
    Frag16 bf[4];
    for (int nt = 0; nt < 4; ++nt) {
      const _Float16* pb = sB + (nt * 16 + mrow) * 136 + kc + hh * 8;
      bf[nt].h[0] = *(const v8h*)pb;
      bf[nt].h[1] = *(const v8h*)(pb + 16);
    }
    for (int nt = 0; nt < 4; ++nt)
      acc[nt] = __builtin_amdgcn_wmma_f32_16x16x32_f16(
          false, a.v, false, bf[nt].v, (short)0, acc[nt], false, false);
  }
  for (int nt = 0; nt < 4; ++nt) {
    int n = nt * 16 + mrow;
    float s = g[n] * rsqrtf(vr[n] + EPSF);
    for (int r = 0; r < 8; ++r) {
      int m = wave * 16 + r + 8 * hh;
      int p = p0 + m; int b = p / 1920; int rr = p % 1920; int y = rr / 80; int x = rr % 80;
      float v = (acc[nt][r] - mn[n]) * s + bt[n];
      out[((b * 64 + n) * 24 + y) * 80 + x] = eluf(v);
    }
  }
}

// ---------------------------------------------------------------------------
// K7: WMMA implicit-GEMM fusion conv, double-buffered LDS with TDM weight
// prefetch (issue tap t+1, s_wait_tensorcnt(1) -> tap t resident), followed
// by a second WMMA GEMM for the outf 1x1 slice accumulated into d_out.
// ---------------------------------------------------------------------------
__device__ __forceinline__ void stage_a_tile(const float* __restrict__ branch,
                                             const float* __restrict__ fid,
                                             int b, int y, int x0, int tap, int tid,
                                             _Float16* __restrict__ dst) {
  int dy = tap / 3, dx = tap % 3;
  int yy = y + dy - 1;
  for (int i = 0; i < 64; ++i) {
    int fl = i * 128 + tid; int px = fl & 63; int ch = fl >> 6;
    int xx = x0 + px + dx - 1;
    float v = 0.f;
    if (yy >= 0 && yy < 96 && xx >= 0 && xx < 320) {
      if (ch < 64) v = branch[((b * 64 + ch) * 24 + (yy >> 2)) * 80 + (xx >> 2)];
      else         v = fid[((b * 64 + (ch - 64)) * 96 + yy) * 320 + xx];
    }
    dst[px * 136 + ch] = (_Float16)v;
  }
}

__global__ __launch_bounds__(128) void k_fusion(PtrTab T, const float* __restrict__ branch,
                                                const _Float16* __restrict__ wbuf,
                                                int br, float* __restrict__ out) {
  __shared__ _Float16 sA[2][64 * 136];
  __shared__ _Float16 sB[2][64 * 136];
  __shared__ _Float16 sC[64 * 72];
  const float* fid = T.p[2 + br];
  const float* fb  = T.p[145 + br * 6];
  const float* g   = T.p[146 + br * 6];
  const float* bt  = T.p[147 + br * 6];
  const float* mn  = T.p[148 + br * 6];
  const float* vr  = T.p[149 + br * 6];
  const float* ofb = T.p[157];
#if !HAVE_TDM
  const float* fw  = T.p[144 + br * 6];
  const float* ofw = T.p[156];
#endif

  int bid = blockIdx.x;
  int xt = bid % 5; int y = (bid / 5) % 96; int b = bid / (5 * 96);
  int x0 = xt * 64;
  int tid = threadIdx.x;
  int wave = tid >> 5, lane = tid & 31, mrow = lane & 15, hh = lane >> 4;

  v8f acc[4];
  for (int i = 0; i < 4; ++i)
    for (int j = 0; j < 8; ++j) acc[i][j] = 0.f;

  // prologue: stage A(0); kick off TDM for tap 0 weights
  stage_a_tile(branch, fid, b, y, x0, 0, tid, sA[0]);
#if HAVE_TDM
  if (tid == 0) tdm_load_2d_f16(wbuf + (br * 9 + 0) * 8192, sB[0], 64, 128, 5, 3);
#endif

  for (int tap = 0; tap < 9; ++tap) {
    int cur = tap & 1, nxt = cur ^ 1;
    __syncthreads();  // closes prior reads of the nxt buffers; publishes sA[cur]
#if HAVE_TDM
    if (tid == 0) {
      if (tap < 8) {
        tdm_load_2d_f16(wbuf + (br * 9 + tap + 1) * 8192, sB[nxt], 64, 128, 5, 3);
        __builtin_amdgcn_s_wait_tensorcnt(1);  // tap's DMA done; tap+1 in flight
      } else {
        __builtin_amdgcn_s_wait_tensorcnt(0);
      }
    }
#else
    {
      int dy = tap / 3, dx = tap % 3;
      for (int i = 0; i < 64; ++i) {
        int fl = i * 128 + tid; int n = fl >> 7; int k = fl & 127;
        sB[cur][n * 136 + k] = (_Float16)fw[((n * 128 + k) * 3 + dy) * 3 + dx];
      }
    }
#endif
    __syncthreads();  // publishes sB[cur]
    for (int kc = 0; kc < 128; kc += 32) {
      Frag16 a;
      const _Float16* pa = sA[cur] + (wave * 16 + mrow) * 136 + kc + hh * 8;
      a.h[0] = *(const v8h*)pa;
      a.h[1] = *(const v8h*)(pa + 16);
      Frag16 bf[4];
      for (int nt = 0; nt < 4; ++nt) {
        const _Float16* pb = sB[cur] + (nt * 16 + mrow) * 136 + kc + hh * 8;
        bf[nt].h[0] = *(const v8h*)pb;
        bf[nt].h[1] = *(const v8h*)(pb + 16);
      }
      for (int nt = 0; nt < 4; ++nt)
        acc[nt] = __builtin_amdgcn_wmma_f32_16x16x32_f16(
            false, a.v, false, bf[nt].v, (short)0, acc[nt], false, false);
    }
    if (tap < 8) stage_a_tile(branch, fid, b, y, x0, tap + 1, tid, sA[nxt]);
  }
  __syncthreads();  // close final tile reads before sB reuse

  // epilogue 1: bias + BN + ELU -> sC (f16)
  for (int nt = 0; nt < 4; ++nt) {
    int n = nt * 16 + mrow;
    float sc = g[n] * rsqrtf(vr[n] + EPSF);
    for (int r = 0; r < 8; ++r) {
      int m = wave * 16 + r + 8 * hh;
      float v = acc[nt][r] + fb[n];
      v = (v - mn[n]) * sc + bt[n];
      sC[m * 72 + n] = (_Float16)eluf(v);
    }
  }
#if HAVE_TDM
  if (tid == 0) {
    tdm_load_2d_f16(wbuf + 147456 + br * 4096, sB[0], 64, 64, 4, 35);
    __builtin_amdgcn_s_wait_tensorcnt(0);
  }
  __syncthreads();
#else
  __syncthreads();
  for (int i = 0; i < 32; ++i) {
    int fl = i * 128 + tid; int n = fl >> 6; int k = fl & 63;
    sB[0][n * 136 + k] = (_Float16)ofw[n * 128 + br * 64 + k];
  }
  __syncthreads();
#endif

  v8f acc2[4];
  for (int i = 0; i < 4; ++i)
    for (int j = 0; j < 8; ++j) acc2[i][j] = 0.f;
  for (int kc = 0; kc < 64; kc += 32) {
    Frag16 a;
    const _Float16* pa = sC + (wave * 16 + mrow) * 72 + kc + hh * 8;
    a.h[0] = *(const v8h*)pa;
    a.h[1] = *(const v8h*)(pa + 16);
    Frag16 bf[4];
    for (int nt = 0; nt < 4; ++nt) {
      const _Float16* pb = sB[0] + (nt * 16 + mrow) * 136 + kc + hh * 8;
      bf[nt].h[0] = *(const v8h*)pb;
      bf[nt].h[1] = *(const v8h*)(pb + 16);
    }
    for (int nt = 0; nt < 4; ++nt)
      acc2[nt] = __builtin_amdgcn_wmma_f32_16x16x32_f16(
          false, a.v, false, bf[nt].v, (short)0, acc2[nt], false, false);
  }
  for (int nt = 0; nt < 4; ++nt) {
    int n = nt * 16 + mrow;
    for (int r = 0; r < 8; ++r) {
      int m = wave * 16 + r + 8 * hh;
      int oidx = ((b * 64 + n) * 96 + y) * 320 + x0 + m;
      if (br == 0) {
        out[oidx] = acc2[nt][r];
      } else {
        float v = out[oidx] + acc2[nt][r] + ofb[n];
        out[oidx] = eluf(v);
      }
    }
  }
}

// ---------------------------------------------------------------------------
extern "C" void kernel_launch(void* const* d_in, const int* in_sizes, int n_in,
                              void* d_out, int out_size, void* d_ws, size_t ws_size,
                              hipStream_t stream) {
  (void)in_sizes; (void)out_size; (void)ws_size;
  PtrTab T;
  for (int i = 0; i < 164; ++i) T.p[i] = (i < n_in) ? (const float*)d_in[i] : nullptr;

  float* ws       = (float*)d_ws;
  float* buf_d1   = ws;                    // 8,847,360
  float* buf_d2   = buf_d1 + 8847360;      // 2,211,840
  float* buf_emb  = buf_d2 + 2211840;      // 8,847,360
  float* buf_din  = buf_emb + 8847360;     // 2,949,120
  float* buf_dout = buf_din + 2949120;     // 2,949,120
  float* buf_br   = buf_dout + 2949120;    // 1,474,560
  _Float16* wbuf  = (_Float16*)(buf_br + 1474560); // 163,840 f16 weight tiles
  float* out = (float*)d_out;

  k_prep    <<<640,   256, 0, stream>>>(T, wbuf);
  k_down1   <<<34560, 256, 0, stream>>>(T, buf_d1);
  k_down2   <<<8640,  256, 0, stream>>>(T, buf_d1, buf_d2);
  k_emb     <<<34560, 256, 0, stream>>>(T, buf_d2, buf_emb);
  k_mamba   <<<5760,  256, 0, stream>>>(T, buf_emb, buf_din);
  k_dec     <<<11520, 256, 0, stream>>>(T, buf_din, buf_dout);
  k_branch_w<<<360,   128, 0, stream>>>(T, buf_dout, wbuf, 0, buf_br);
  k_fusion  <<<5760,  128, 0, stream>>>(T, buf_br, wbuf, 0, out);
  k_branch_w<<<360,   128, 0, stream>>>(T, buf_dout, wbuf, 1, buf_br);
  k_fusion  <<<5760,  128, 0, stream>>>(T, buf_br, wbuf, 1, out);
}
